// ClusteringLayer_3384434229450
// MI455X (gfx1250) — matrix operand order repaired and checked by reading, not compiled, not run on hardware
//
#include <hip/hip_runtime.h>

typedef __attribute__((ext_vector_type(16))) __bf16 v16bf;
typedef __attribute__((ext_vector_type(8)))  float  v8f;
typedef __attribute__((ext_vector_type(4)))  unsigned int v4u;
typedef __attribute__((ext_vector_type(8)))  int v8i;
typedef __attribute__((ext_vector_type(4)))  int v4i;

#define DIMS   256   // K
#define NCLUST 512   // N
#define NTH    16    // N tiles per wave (half of 32)
#define KC     8     // K chunks of 32

__device__ __forceinline__ unsigned int f2bf_rn(float f) {
  unsigned int u = __builtin_bit_cast(unsigned int, f);
  return ((u + 0x7FFFu + ((u >> 16) & 1u)) >> 16) & 0xFFFFu;
}
__device__ __forceinline__ float bf2f(unsigned int h) {
  return __builtin_bit_cast(float, h << 16);
}

union FragU { unsigned int u[8]; uint4 q[2]; v16bf v; };

// Tensor DMA: one contiguous 64 KB line (16384 dwords) global -> LDS
__device__ __forceinline__ void tdm_load_chunk(const unsigned* gsrc,
                                               unsigned ldsAddr) {
  unsigned long long ga = (unsigned long long)(uintptr_t)gsrc;
  v4u g0;
  g0[0] = 1u;                                        // count=1, user mode
  g0[1] = ldsAddr;                                   // lds_addr
  g0[2] = (unsigned)ga;                              // global_addr[31:0]
  g0[3] = (unsigned)((ga >> 32) & 0x01FFFFFFull)     // global_addr[56:32]
          | (2u << 30);                              // type=2 (image)
  v8i g1;
  g1[0] = (int)(2u << 16);        // data_size=4B; no mask/flags
  g1[1] = (int)(16384u << 16);    // tensor_dim0[15:0]=16384 in [31:16]
  g1[2] = (int)(1u << 16);        // tensor_dim0 hi=0; tensor_dim1=1
  g1[3] = (int)(16384u << 16);    // tensor_dim1 hi=0; tile_dim0=16384
  g1[4] = 1;                      // tile_dim1=1; tile_dim2=0
  g1[5] = 16384;                  // tensor_dim0_stride[31:0]
  g1[6] = 0;
  g1[7] = 0;
  v4i z4 = {0, 0, 0, 0};
#if defined(__clang_major__) && __clang_major__ >= 23
  v8i z8 = {0, 0, 0, 0, 0, 0, 0, 0};
  __builtin_amdgcn_tensor_load_to_lds(g0, g1, z4, z4, z8, 0);
#else
  __builtin_amdgcn_tensor_load_to_lds(g0, g1, z4, z4, 0);
#endif
}

// ---------------------------------------------------------------------------
// Prep 1: clusters f32 [512,256] -> bf16 hi/lo dwords, kc-contiguous so one
// TDM transfer stages a whole K-chunk (64 KB) into LDS:
//   ws dword layout: [kc][ hi: 8192 dwords | lo: 8192 dwords ]
//   within hi/lo: dword = (t*32 + lane)*8 + p
//     column n = t*16 + (lane&15)
//     K pair   = kc*32 + (lane>=16 ? 8 : 0) + (p>=4 ? 16 : 0) + 2(p&3)  (even K in low 16 bits)
// ---------------------------------------------------------------------------
__global__ void prep_pack(const float* __restrict__ clusters,
                          unsigned int* __restrict__ ws) {
  int idx  = blockIdx.x * 256 + threadIdx.x;   // 0..65535
  int p    = idx & 7;
  int lane = (idx >> 3) & 31;
  int t    = (idx >> 8) & 31;
  int kc   = idx >> 13;
  int rem  = idx & 8191;
  int n  = t * 16 + (lane & 15);
  int k  = kc * 32 + ((lane & 16) ? 8 : 0) + ((p & 4) ? 16 : 0) + ((p & 3) << 1);
  float c0 = clusters[n * DIMS + k];
  float c1 = clusters[n * DIMS + k + 1];
  unsigned int h0 = f2bf_rn(c0), h1 = f2bf_rn(c1);
  unsigned int l0 = f2bf_rn(c0 - bf2f(h0)), l1 = f2bf_rn(c1 - bf2f(h1));
  ws[kc * 16384 + rem]        = h0 | (h1 << 16);
  ws[kc * 16384 + 8192 + rem] = l0 | (l1 << 16);
}

// Prep 2: |c_k|^2 per cluster
__global__ void prep_csq(const float* __restrict__ clusters,
                         float* __restrict__ csq) {
  int n = blockIdx.x * 256 + threadIdx.x;
  if (n < NCLUST) {
    const float4* r = (const float4*)(clusters + (size_t)n * DIMS);
    float s = 0.f;
    for (int j = 0; j < DIMS / 4; ++j) {
      float4 v = r[j];
      s += v.x * v.x + v.y * v.y + v.z * v.z + v.w * v.w;
    }
    csq[n] = s;
  }
}

// ---------------------------------------------------------------------------
// Main kernel: 256 threads = 8 waves = 4 row-groups x 2 N-halves.
// Each wave: 16 rows x 256 cols (16 N-tiles -> 128 acc VGPRs).
// B chunks stream through a double-buffered 2x64KB LDS window via the
// Tensor Data Mover, overlapped with WMMA compute.
// ---------------------------------------------------------------------------
__launch_bounds__(256)
__global__ void dec_q_kernel(const float* __restrict__ x,
                             const unsigned int* __restrict__ wsb,
                             const float* __restrict__ csq,
                             float* __restrict__ out) {
  // 128 KB: two buffers; each: hi fragments [uint4 0..2047], lo [2048..4095]
  __shared__ uint4 ldsB[8192];

  const int tid    = threadIdx.x;
  const int lane   = tid & 31;
  const int wave   = tid >> 5;
  const int rgrp   = wave >> 1;       // row group 0..3
  const int nhalf  = wave & 1;        // N half 0..1
  const int rowBase = blockIdx.x * 64 + rgrp * 16;
  const int m      = rowBase + (lane & 15);   // A fragment row for this lane
  const int kbase  = (lane & 16) ? 8 : 0;     // A fragment K base for this lane
  const int tBase  = nhalf * NTH;             // first global N tile for this wave

  v8f acc[NTH];
  #pragma unroll
  for (int lt = 0; lt < NTH; ++lt)
    acc[lt] = (v8f){0.f, 0.f, 0.f, 0.f, 0.f, 0.f, 0.f, 0.f};

  float xsp = 0.f;  // partial |x_m|^2 over this lane's K subset

  const unsigned ldsOff = (unsigned)(uintptr_t)&ldsB[0];

  // Prime the pipeline: chunk 0 -> buffer 0
  if (wave == 0) tdm_load_chunk(wsb, ldsOff);

  for (int kc = 0; kc < KC; ++kc) {
    if (wave == 0) {
      if (kc < KC - 1) {
        // stream next chunk into the other buffer, behind this chunk's compute
        tdm_load_chunk(wsb + (size_t)(kc + 1) * 16384,
                       ldsOff + ((kc + 1) & 1) * 65536u);
        __builtin_amdgcn_s_wait_tensorcnt(1);  // chunk kc resident
      } else {
        __builtin_amdgcn_s_wait_tensorcnt(0);  // last chunk resident
      }
    }
    __syncthreads();   // chunk kc visible to all waves

    // A fragment: lane covers K = {kbase..kbase+7, kbase+16..kbase+23}
    const float* xr = x + (size_t)m * DIMS + kc * 32 + kbase;
    float4 r0 = ((const float4*)xr)[0];
    float4 r1 = ((const float4*)xr)[1];
    float4 r2 = ((const float4*)(xr + 16))[0];
    float4 r3 = ((const float4*)(xr + 16))[1];
    float f[16] = {r0.x, r0.y, r0.z, r0.w, r1.x, r1.y, r1.z, r1.w,
                   r2.x, r2.y, r2.z, r2.w, r3.x, r3.y, r3.z, r3.w};
    FragU ah, al;
    #pragma unroll
    for (int d = 0; d < 8; ++d) {
      float a0 = f[2 * d], a1 = f[2 * d + 1];
      xsp += a0 * a0 + a1 * a1;
      unsigned int h0 = f2bf_rn(a0), h1 = f2bf_rn(a1);
      unsigned int l0 = f2bf_rn(a0 - bf2f(h0)), l1 = f2bf_rn(a1 - bf2f(h1));
      ah.u[d] = h0 | (h1 << 16);
      al.u[d] = l0 | (l1 << 16);
    }

    const int bufBase = (kc & 1) * 4096;   // uint4 index of active buffer
    #pragma unroll
    for (int lt = 0; lt < NTH; ++lt) {
      FragU bh, bl;
      int bi = bufBase + ((tBase + lt) * 32 + lane) * 2;
      bh.q[0] = ldsB[bi];
      bh.q[1] = ldsB[bi + 1];
      bl.q[0] = ldsB[bi + 2048 * 2];       // lo region: +2048 uint4
      bl.q[1] = ldsB[bi + 2048 * 2 + 1];
      // split-precision: hi*hi + hi*lo + lo*hi  (~fp32 accuracy)
      acc[lt] = __builtin_amdgcn_wmma_f32_16x16x32_bf16(
                    false, ah.v, false, bh.v, (short)0, acc[lt], false, false);
      acc[lt] = __builtin_amdgcn_wmma_f32_16x16x32_bf16(
                    false, ah.v, false, bl.v, (short)0, acc[lt], false, false);
      acc[lt] = __builtin_amdgcn_wmma_f32_16x16x32_bf16(
                    false, al.v, false, bh.v, (short)0, acc[lt], false, false);
    }
    __syncthreads();   // buffer (kc&1) consumed; may be refilled next iteration
  }

  // |x_m|^2: lanes l and l^16 hold complementary K halves of row (l&15)
  xsp += __shfl_xor(xsp, 16, 32);
  const int hsel = (lane & 16) ? 8 : 0;  // C/D layout: lanes>=16 hold rows M=v+8
  float xsqv[8];
  #pragma unroll
  for (int v = 0; v < 8; ++v) xsqv[v] = __shfl(xsp, v + hsel, 32);

  float rs[8];
  #pragma unroll
  for (int v = 0; v < 8; ++v) rs[v] = 0.f;

  // q = 1/(1+d); accumulate row sums over this wave's N half
  #pragma unroll
  for (int lt = 0; lt < NTH; ++lt) {
    float cn = csq[(tBase + lt) * 16 + (lane & 15)];
    #pragma unroll
    for (int v = 0; v < 8; ++v) {
      float d  = xsqv[v] + cn - 2.0f * acc[lt][v];
      float qv = __builtin_amdgcn_rcpf(1.0f + d);
      acc[lt][v] = qv;
      rs[v] += qv;
    }
  }

  // reduce across the 16 lanes holding this half's columns
  #pragma unroll
  for (int v = 0; v < 8; ++v) {
    rs[v] += __shfl_xor(rs[v], 1, 32);
    rs[v] += __shfl_xor(rs[v], 2, 32);
    rs[v] += __shfl_xor(rs[v], 4, 32);
    rs[v] += __shfl_xor(rs[v], 8, 32);
  }

  // combine the two N-halves via LDS (reuse ldsB storage; B no longer needed)
  __syncthreads();
  float* rsLDS = (float*)ldsB;   // [wave][row] : 8 x 16 floats
  if ((lane & 15) == 0) {
    #pragma unroll
    for (int v = 0; v < 8; ++v) rsLDS[wave * 16 + v + hsel] = rs[v];
  }
  __syncthreads();
  const int partner = wave ^ 1;
  float inv[8];
  #pragma unroll
  for (int v = 0; v < 8; ++v) {
    float tot = rs[v] + rsLDS[partner * 16 + v + hsel];
    inv[v] = __builtin_amdgcn_rcpf(tot);
  }

  // normalized store (non-temporal: output is never re-read)
  #pragma unroll
  for (int lt = 0; lt < NTH; ++lt) {
    int ncol = (tBase + lt) * 16 + (lane & 15);
    #pragma unroll
    for (int v = 0; v < 8; ++v) {
      int row = rowBase + v + hsel;
      __builtin_nontemporal_store(acc[lt][v] * inv[v],
                                  out + (size_t)row * NCLUST + ncol);
    }
  }
}

// ---------------------------------------------------------------------------
extern "C" void kernel_launch(void* const* d_in, const int* in_sizes, int n_in,
                              void* d_out, int out_size, void* d_ws, size_t ws_size,
                              hipStream_t stream) {
  (void)in_sizes; (void)n_in; (void)out_size; (void)ws_size;
  const float* x        = (const float*)d_in[0];   // [65536, 256] f32
  const float* clusters = (const float*)d_in[1];   // [512, 256]  f32
  float* out = (float*)d_out;                      // [65536, 512] f32

  // workspace: 512 KB packed bf16 hi/lo (kc-contiguous) + 2 KB csq
  unsigned int* ws  = (unsigned int*)d_ws;
  float*        csq = (float*)(ws + 8 * 16384);

  prep_pack<<<256, 256, 0, stream>>>(clusters, ws);
  prep_csq<<<2, 256, 0, stream>>>(clusters, csq);
  dec_q_kernel<<<1024, 256, 0, stream>>>(x, ws, csq, out);
}
